// ScaleDotProductAttention_43069932044519
// MI455X (gfx1250) — matrix operand order repaired and checked by reading, not compiled, not run on hardware
//
#include <hip/hip_runtime.h>
#include <hip/hip_bf16.h>

typedef __attribute__((ext_vector_type(16))) __bf16 v16bf;
typedef __attribute__((ext_vector_type(8)))  float  v8f;

// Problem constants (from reference): [B,H,S,D] = [4,16,2048,128]
#define ATT_S 2048
#define ATT_D 128
#define QTILE 128        // q rows per workgroup (8 waves x 16 rows)
#define KTILE 64         // key rows per flash iteration
#define NBLK  (ATT_S / KTILE)
// LDS strides in DWORDS (dword = 2 bf16)
#define KSTRW 68         // K row stride: 128 bf16 = 64 dw + 4 pad
#define VSTRW 36         // V^T row stride: 64 bf16 = 32 dw + 4 pad
#define PSTRW 36         // P row stride:  64 bf16 = 32 dw + 4 pad
// Dynamic LDS carve-up (dwords)
#define KB_DW (KTILE * KSTRW)          // 4352 per buffer
#define VT_DW (ATT_D * VSTRW)          // 4608 per buffer
#define PB_DW (16 * PSTRW)             // 576 per wave
#define SMEM_DW (2 * KB_DW + 2 * VT_DW + 8 * PB_DW)   // 22528 dw = 90112 B

union Frag16 { uint4 q[2]; v16bf v; };

static __device__ __forceinline__ unsigned pack_bf16(float a, float b) {
  union { __bf16 h[2]; unsigned u; } t;
  t.h[0] = (__bf16)a;
  t.h[1] = (__bf16)b;
  return t.u;                       // v_cvt_pk_bf16_f32
}

// XOR-butterfly within each 16-lane row via v_permlane16_b32 (VALU, no LDS)
static __device__ __forceinline__ float pl16_f(float x, int s1, int s2) {
  int xi = __builtin_bit_cast(int, x);
  int r  = __builtin_amdgcn_permlane16(xi, xi, s1, s2, false, false);
  return __builtin_bit_cast(float, r);
}
static __device__ __forceinline__ float rowmax16(float x) {
  x = fmaxf(x, pl16_f(x, (int)0x67452301u, (int)0xEFCDAB89u));  // xor 1
  x = fmaxf(x, pl16_f(x, (int)0x54761032u, (int)0xDCFE98BAu));  // xor 2
  x = fmaxf(x, pl16_f(x, (int)0x32107654u, (int)0xBA98FEDCu));  // xor 4
  x = fmaxf(x, pl16_f(x, (int)0xFEDCBA98u, (int)0x76543210u));  // xor 8
  return x;
}
static __device__ __forceinline__ float rowsum16(float x) {
  x += pl16_f(x, (int)0x67452301u, (int)0xEFCDAB89u);  // xor 1
  x += pl16_f(x, (int)0x54761032u, (int)0xDCFE98BAu);  // xor 2
  x += pl16_f(x, (int)0x32107654u, (int)0xBA98FEDCu);  // xor 4
  x += pl16_f(x, (int)0xFEDCBA98u, (int)0x76543210u);  // xor 8
  return x;
}

__global__ __launch_bounds__(256, 4)   // <=256 VGPRs: 2 workgroups/WGP
void fa_fwd_bf16wmma(const float* __restrict__ q,
                     const float* __restrict__ k,
                     const float* __restrict__ v,
                     float* __restrict__ out) {
  extern __shared__ unsigned smem[];
  unsigned* sKb = smem;                         // [2][KTILE][KSTRW]
  unsigned* sVT = smem + 2 * KB_DW;             // [2][ATT_D][VSTRW]
  unsigned* sPb = smem + 2 * KB_DW + 2 * VT_DW; // [8][16][PSTRW]

  const int tid  = threadIdx.x;
  const int wave = tid >> 5;
  const int lane = tid & 31;
  const int half = lane >> 4;
  const int nn   = lane & 15;

  const int bh    = blockIdx.y;
  const int qrow0 = blockIdx.x * QTILE + wave * 16;

  const float scale = 0.08838834764831845f;   // 1/sqrt(128)

  // ---- Q as 4 bf16 A-fragments (16x32), scale folded in ----
  v16bf qa[4];
  {
    const float* qbase = q + ((size_t)bh * ATT_S + (size_t)(qrow0 + nn)) * ATT_D;
#pragma unroll
    for (int f = 0; f < 4; ++f) {
#pragma unroll
      for (int vv = 0; vv < 8; ++vv) {
        int d = f * 32 + ((vv < 4) ? 0 : 16) + half * 8 + (vv & 3) * 2;
        float2 t = *(const float2*)(qbase + d);
        qa[f][2 * vv]     = (__bf16)(t.x * scale);
        qa[f][2 * vv + 1] = (__bf16)(t.y * scale);
      }
    }
  }

  float m_run[8], l_run[8];
  v8f   o[8];
#pragma unroll
  for (int r = 0; r < 8; ++r) { m_run[r] = -1.0e30f; l_run[r] = 0.0f; }
#pragma unroll
  for (int nb = 0; nb < 8; ++nb) o[nb] = (v8f){};

  const float* kbh = k + (size_t)bh * ATT_S * ATT_D;
  const float* vbh = v + (size_t)bh * ATT_S * ATT_D;

  // ---- staging (K and V phases never overlap in register liveness) ----
  float4 kreg[8];                    // K: 2048 float4 tasks
  float4 vregA[4], vregB[4];         // V: 1024 row-pair float4 tasks

  auto stage_load_K = [&](int jb) {
    const float4* kg = (const float4*)(kbh + (size_t)jb * KTILE * ATT_D);
#pragma unroll
    for (int it = 0; it < 8; ++it) kreg[it] = kg[tid + it * 256];
  };
  auto stage_store_K = [&](int buf) {
    unsigned* kd = sKb + buf * KB_DW;
#pragma unroll
    for (int it = 0; it < 8; ++it) {
      int i = tid + it * 256;
      int r = i >> 5, c4 = i & 31;
      uint2 pk;
      pk.x = pack_bf16(kreg[it].x, kreg[it].y);
      pk.y = pack_bf16(kreg[it].z, kreg[it].w);
      *(uint2*)(&kd[r * KSTRW + c4 * 2]) = pk;
    }
  };
  auto stage_load_V = [&](int jb) {
    const float* vg = vbh + (size_t)jb * KTILE * ATT_D;
#pragma unroll
    for (int it = 0; it < 4; ++it) {
      int id = tid + it * 256;        // task: rp = id>>5 (0..31), c4 = id&31
      int rp = id >> 5, c4 = id & 31;
      vregA[it] = *(const float4*)(vg + (2 * rp) * ATT_D + 4 * c4);
      vregB[it] = *(const float4*)(vg + (2 * rp + 1) * ATT_D + 4 * c4);
    }
  };
  auto stage_store_V = [&](int buf) {
    unsigned* vd = sVT + buf * VT_DW;
#pragma unroll
    for (int it = 0; it < 4; ++it) {
      int id = tid + it * 256;
      int rp = id >> 5, c4 = id & 31;
      vd[(4 * c4 + 0) * VSTRW + rp] = pack_bf16(vregA[it].x, vregB[it].x);
      vd[(4 * c4 + 1) * VSTRW + rp] = pack_bf16(vregA[it].y, vregB[it].y);
      vd[(4 * c4 + 2) * VSTRW + rp] = pack_bf16(vregA[it].z, vregB[it].z);
      vd[(4 * c4 + 3) * VSTRW + rp] = pack_bf16(vregA[it].w, vregB[it].w);
    }
  };

  stage_load_K(0);
  stage_store_K(0);
  stage_load_V(0);
  stage_store_V(0);
  __syncthreads();

  for (int jb = 0; jb < NBLK; ++jb) {
    const int buf = jb & 1;
    const unsigned* kd = sKb + buf * KB_DW;
    const unsigned* vd = sVT + buf * VT_DW;

    if (jb + 2 < NBLK) {   // L2 prefetch two tiles ahead: 256 thr x 128B = full tile
      __builtin_prefetch(kbh + (size_t)(jb + 2) * KTILE * ATT_D + tid * 32, 0, 1);
      __builtin_prefetch(vbh + (size_t)(jb + 2) * KTILE * ATT_D + tid * 32, 0, 1);
    }

    // ---- S = Q K^T : four 16x16 score blocks, frag loads pipelined 1 ahead ----
    v8f sc[4];
    {
      Frag16 fr[2];
      auto loadK = [&](Frag16& dst, int idx) {
        int kb = idx >> 2, f = idx & 3;
        const unsigned* p = &kd[(kb * 16 + nn) * KSTRW + f * 16 + half * 4];
        dst.q[0] = *(const uint4*)(p);
        dst.q[1] = *(const uint4*)(p + 8);
      };
      loadK(fr[0], 0);
#pragma unroll
      for (int idx = 0; idx < 16; ++idx) {
        int kb = idx >> 2, f = idx & 3;
        if (idx < 15) loadK(fr[(idx + 1) & 1], idx + 1);
        v8f c = (f == 0) ? (v8f){} : sc[kb];
        sc[kb] = __builtin_amdgcn_wmma_f32_16x16x32_bf16(
            false, qa[f], false, fr[idx & 1].v, (short)0, c, false, false);
      }
    }

    // ---- Online softmax over 64 keys (f32 denominator, bf16 numerator) ----
    float scale_r[8];
    __bf16* pW = (__bf16*)&sPb[wave * PB_DW];
#pragma unroll
    for (int r = 0; r < 8; ++r) {
      float bm = fmaxf(fmaxf(sc[0][r], sc[1][r]), fmaxf(sc[2][r], sc[3][r]));
      bm = rowmax16(bm);                       // v_permlane16 butterfly
      float mnew = fmaxf(m_run[r], bm);
      float so   = __expf(m_run[r] - mnew);
      m_run[r]   = mnew;
      scale_r[r] = so;
      float p0 = __expf(sc[0][r] - mnew);
      float p1 = __expf(sc[1][r] - mnew);
      float p2 = __expf(sc[2][r] - mnew);
      float p3 = __expf(sc[3][r] - mnew);
      pW[(half * 8 + r) * (2 * PSTRW) +  0 + nn] = (__bf16)p0;
      pW[(half * 8 + r) * (2 * PSTRW) + 16 + nn] = (__bf16)p1;
      pW[(half * 8 + r) * (2 * PSTRW) + 32 + nn] = (__bf16)p2;
      pW[(half * 8 + r) * (2 * PSTRW) + 48 + nn] = (__bf16)p3;
      float ps = (p0 + p1) + (p2 + p3);
      ps = rowsum16(ps);                       // f32 row sum across 16 lanes
      l_run[r] = l_run[r] * so + ps;           // l state closed out here
    }

    // Rescale O accumulators (scale_r dies here)
#pragma unroll
    for (int nb = 0; nb < 8; ++nb)
#pragma unroll
      for (int r = 0; r < 8; ++r) o[nb][r] *= scale_r[r];

    // DS in-order per wave: fence so P reads see all 32 lanes' stores
    asm volatile("s_wait_dscnt 0" ::: "memory");

    // ---- P in A layout: two 16x32 fragments (key chunks) ----
    Frag16 pa[2];
#pragma unroll
    for (int c = 0; c < 2; ++c) {
      const unsigned* p = &sPb[wave * PB_DW + nn * PSTRW + c * 16 + half * 4];
      pa[c].q[0] = *(const uint4*)(p);
      pa[c].q[1] = *(const uint4*)(p + 8);
    }

    // ---- O += P V, interleaved with time-sliced staging of tile jb+1 ----
    Frag16 fr[2];
    auto loadV = [&](Frag16& dst, int c, int nb) {
      const unsigned* p = &vd[(nb * 16 + nn) * VSTRW + c * 16 + half * 4];
      dst.q[0] = *(const uint4*)(p);
      dst.q[1] = *(const uint4*)(p + 8);
    };

    if (jb + 1 < NBLK) stage_load_K(jb + 1);   // kreg born (32 VGPRs)

    loadV(fr[0], 0, 0);
#pragma unroll
    for (int nb = 0; nb < 8; ++nb) {           // PV chunk 0: covers K load latency
      if (nb < 7) loadV(fr[(nb + 1) & 1], 0, nb + 1);
      o[nb] = __builtin_amdgcn_wmma_f32_16x16x32_bf16(
          false, pa[0].v, false, fr[nb & 1].v, (short)0, o[nb], false, false);
    }

    if (jb + 1 < NBLK) {
      stage_store_K((jb + 1) & 1);             // kreg dies
      stage_load_V(jb + 1);                    // vreg born (32 VGPRs)
    }

    loadV(fr[0], 1, 0);
#pragma unroll
    for (int nb = 0; nb < 8; ++nb) {           // PV chunk 1: covers V load latency
      if (nb < 7) loadV(fr[(nb + 1) & 1], 1, nb + 1);
      o[nb] = __builtin_amdgcn_wmma_f32_16x16x32_bf16(
          false, pa[1].v, false, fr[nb & 1].v, (short)0, o[nb], false, false);
    }

    if (jb + 1 < NBLK) stage_store_V((jb + 1) & 1);   // vreg dies
    __syncthreads();
  }

  // ---- Normalize and store ----
#pragma unroll
  for (int r = 0; r < 8; ++r) {
    float inv = 1.0f / l_run[r];
    float* orow = out + ((size_t)bh * ATT_S + (size_t)(qrow0 + half * 8 + r)) * ATT_D;
#pragma unroll
    for (int nb = 0; nb < 8; ++nb) {
      orow[nb * 16 + nn] = o[nb][r] * inv;
    }
  }
}

extern "C" void kernel_launch(void* const* d_in, const int* in_sizes, int n_in,
                              void* d_out, int out_size, void* d_ws, size_t ws_size,
                              hipStream_t stream) {
  (void)in_sizes; (void)n_in; (void)d_ws; (void)ws_size; (void)out_size;
  const float* q = (const float*)d_in[0];
  const float* k = (const float*)d_in[1];
  const float* v = (const float*)d_in[2];
  float* out = (float*)d_out;

  const size_t smem_bytes = (size_t)SMEM_DW * 4;   // 90112 B dynamic LDS
  (void)hipFuncSetAttribute((const void*)fa_fwd_bf16wmma,
                            hipFuncAttributeMaxDynamicSharedMemorySize,
                            (int)smem_bytes);

  dim3 grid(ATT_S / QTILE, 4 * 16);   // 16 q-tiles x (B*H)=64
  dim3 block(256);                    // 8 waves (wave32)
  fa_fwd_bf16wmma<<<grid, block, smem_bytes, stream>>>(q, k, v, out);
}
